// VQGAN_79293686219027
// MI455X (gfx1250) — matrix operand order repaired and checked by reading, not compiled, not run on hardware
//
#include <hip/hip_runtime.h>
#include <hip/hip_bf16.h>
#include <math.h>

typedef __attribute__((ext_vector_type(16))) __bf16 v16bf;
typedef __attribute__((ext_vector_type(8)))  float  v8f;

#define GROUPS 32

// ---------------------------------------------------------------------------
// Weight repack: fp32 OIHW [Cout][Cin][KH*KW]  ->  bf16 [tap][Cout][Cin].
// Makes the WMMA A-operand contiguous in ci and pre-converts to bf16 once.
// ---------------------------------------------------------------------------
__global__ __launch_bounds__(256)
void repack_w_kernel(const float* __restrict__ w, __bf16* __restrict__ wpk,
                     int Cout, int Cin, int KK)
{
    const size_t total = (size_t)Cout * Cin * KK;
    const size_t i = (size_t)blockIdx.x * 256 + threadIdx.x;
    if (i >= total) return;
    const int kk = (int)(i % KK);
    size_t t = i / KK;
    const int ci = (int)(t % Cin);
    const int co = (int)(t / Cin);
    wpk[((size_t)kk * Cout + co) * Cin + ci] = (__bf16)w[i];
}

// ---------------------------------------------------------------------------
// Implicit-GEMM convolution, one wave32 per 32x32 output macro-tile
// (2 Cout-tiles x 2 spatial-tiles, 4 accumulators -> 4 WMMAs per k-chunk,
//  each A/B fragment reused twice; halves L2 re-fetch traffic vs 16x16).
// Tap-outer / channel-inner: for each of the KH*KW taps, K == ci, so
//   B addresses are base + ci*H*W (pure strided adds, no div/mod),
//   A is bf16 [tap][Cout][Cin] (contiguous pairs -> b32 loads),
//   one uniform bounds check per tap per lane per n-tile.
// D = A*B + C via v_wmma_f32_16x16x32_bf16 (fp32 accumulate).
// Bias and optional residual add fused into the epilogue.
// ---------------------------------------------------------------------------
template<int KH, int KW, int STRIDE, int PAD>
__global__ __launch_bounds__(32)
void conv_wmma_kernel(const float* __restrict__ in, const __bf16* __restrict__ wpk,
                      const float* __restrict__ bias, const float* __restrict__ resid,
                      float* __restrict__ out,
                      int B, int Cin, int H, int W, int Cout, int OH, int OW)
{
    const int lane = threadIdx.x;
    const int l15  = lane & 15;
    const int half = lane >> 4;
    const int HWin = H * W;
    const int N    = B * OH * OW;

    // ---- two Cout rows per lane (m-tiles 0/1) ----
    const int  m0   = blockIdx.x * 32 + l15;
    const int  m1   = m0 + 16;
    const bool mok0 = (m0 < Cout);
    const bool mok1 = (m1 < Cout);

    // ---- two spatial columns per lane (n-tiles 0/1) ----
    int nn[2]; bool nok[2]; int ow[2], oh[2], bb[2];
#pragma unroll
    for (int t = 0; t < 2; ++t) {
        nn[t]  = blockIdx.y * 32 + t * 16 + l15;
        nok[t] = (nn[t] < N);
        ow[t] = 0; oh[t] = 0; bb[t] = 0;
        if (nok[t]) { ow[t] = nn[t] % OW; int q = nn[t] / OW; oh[t] = q % OH; bb[t] = q / OH; }
    }

    // Per-lane B-operand channel offsets within a 32-chunk:
    // element e (=2v+j) of the fragment holds k = half*16 + e.
    int boff[16];
#pragma unroll
    for (int e = 0; e < 16; ++e) boff[e] = (half * 16 + e) * HWin;

    // A fragment loader (16-bit A 16x32, element e holds k = half*16 + e)
    auto loadA = [&](const __bf16* wr, bool mok, int c0, bool fullc) -> v16bf {
        union { v16bf h; unsigned u[8]; } a;
        if (mok && fullc) {
            const unsigned* w32 = (const unsigned*)(wr + c0);
#pragma unroll
            for (int v = 0; v < 8; ++v)
                a.u[v] = w32[v + half * 8];
        } else {
#pragma unroll
            for (int v = 0; v < 8; ++v) {
                const int kb = c0 + 2 * v + half * 16;
                a.h[2 * v]     = (mok && kb     < Cin) ? wr[kb]     : (__bf16)0.f;
                a.h[2 * v + 1] = (mok && kb + 1 < Cin) ? wr[kb + 1] : (__bf16)0.f;
            }
        }
        return a.h;
    };

    // B fragment loader (16-bit B 32x16 layout)
    auto loadB = [&](long long ibase, bool valid, int c0, bool fullc) -> v16bf {
        v16bf bf;
        const float* p = in + ibase + (long long)c0 * HWin;
        if (valid && fullc) {
#pragma unroll
            for (int e = 0; e < 16; ++e) bf[e] = (__bf16)p[boff[e]];
        } else {
#pragma unroll
            for (int e = 0; e < 16; ++e) {
                const int ci = c0 + half * 16 + e;
                bf[e] = (valid && ci < Cin) ? (__bf16)p[boff[e]] : (__bf16)0.f;
            }
        }
        return bf;
    };

    v8f acc00 = {}, acc01 = {}, acc10 = {}, acc11 = {};

    for (int tap = 0; tap < KH * KW; ++tap) {
        const int kh = tap / KW, kw = tap % KW;

        bool val[2]; long long ib[2];
#pragma unroll
        for (int t = 0; t < 2; ++t) {
            const int ih = oh[t] * STRIDE + kh - PAD;
            const int iw = ow[t] * STRIDE + kw - PAD;
            val[t] = nok[t] && ih >= 0 && ih < H && iw >= 0 && iw < W;
            ib[t]  = (((long long)bb[t] * Cin) * H + ih) * (long long)W + iw;
        }
        const __bf16* wr0 = wpk + ((size_t)tap * Cout + m0) * Cin;
        const __bf16* wr1 = wr0 + (size_t)16 * Cin;

        for (int c0 = 0; c0 < Cin; c0 += 32) {
            const bool fullc = (c0 + 32) <= Cin;
            const v16bf a0 = loadA(wr0, mok0, c0, fullc);
            const v16bf a1 = loadA(wr1, mok1, c0, fullc);
            const v16bf b0 = loadB(ib[0], val[0], c0, fullc);
            const v16bf b1 = loadB(ib[1], val[1], c0, fullc);

            acc00 = __builtin_amdgcn_wmma_f32_16x16x32_bf16(false, a0, false, b0, (short)0, acc00, false, false);
            acc01 = __builtin_amdgcn_wmma_f32_16x16x32_bf16(false, a0, false, b1, (short)0, acc01, false, false);
            acc10 = __builtin_amdgcn_wmma_f32_16x16x32_bf16(false, a1, false, b0, (short)0, acc10, false, false);
            acc11 = __builtin_amdgcn_wmma_f32_16x16x32_bf16(false, a1, false, b1, (short)0, acc11, false, false);
        }
    }

    // ---- Epilogue: C/D 16x16 layout, lane covers col l15, rows r + half*8 ----
#pragma unroll
    for (int mt = 0; mt < 2; ++mt) {
#pragma unroll
        for (int nt = 0; nt < 2; ++nt) {
            if (!nok[nt]) continue;
            const v8f a = (mt == 0) ? (nt == 0 ? acc00 : acc01)
                                    : (nt == 0 ? acc10 : acc11);
#pragma unroll
            for (int r = 0; r < 8; ++r) {
                const int m = blockIdx.x * 32 + mt * 16 + r + half * 8;
                if (m < Cout) {
                    const size_t o = (((size_t)bb[nt] * Cout + m) * OH + oh[nt]) * OW + ow[nt];
                    float v = a[r] + bias[m];
                    if (resid) v += resid[o];
                    out[o] = v;
                }
            }
        }
    }
}

// ---------------------------------------------------------------------------
// GroupNorm: per-(batch,group) stats, then fused normalize(+SiLU) apply.
// ---------------------------------------------------------------------------
__global__ __launch_bounds__(256)
void gn_stats_kernel(const float* __restrict__ x, float* __restrict__ stats,
                     int C, int HW)
{
    const int bg  = blockIdx.x;               // b*GROUPS + g
    const int cpg = C / GROUPS;
    const int b   = bg / GROUPS, g = bg % GROUPS;
    const size_t base = ((size_t)b * C + (size_t)g * cpg) * HW;
    const size_t n    = (size_t)cpg * HW;

    float s = 0.f, s2 = 0.f;
    for (size_t i = threadIdx.x; i < n; i += 256) {
        float v = x[base + i];
        s += v; s2 += v * v;
    }
    __shared__ float sh[512];
    sh[threadIdx.x] = s; sh[256 + threadIdx.x] = s2;
    __syncthreads();
    for (int st = 128; st > 0; st >>= 1) {
        if (threadIdx.x < st) {
            sh[threadIdx.x]       += sh[threadIdx.x + st];
            sh[256 + threadIdx.x] += sh[256 + threadIdx.x + st];
        }
        __syncthreads();
    }
    if (threadIdx.x == 0) {
        const float inv = 1.f / (float)n;
        const float m   = sh[0] * inv;
        const float var = sh[256] * inv - m * m;
        stats[2 * bg]     = m;
        stats[2 * bg + 1] = rsqrtf(var + 1e-5f);
    }
}

__global__ __launch_bounds__(256)
void gn_apply_kernel(const float* __restrict__ x, const float* __restrict__ stats,
                     const float* __restrict__ gam, const float* __restrict__ bet,
                     float* __restrict__ y, int B, int C, int HW, int do_silu)
{
    const size_t total = (size_t)B * C * HW;
    const size_t i = (size_t)blockIdx.x * 256 + threadIdx.x;
    if (i >= total) return;
    const int c = (int)((i / HW) % C);
    const int b = (int)(i / ((size_t)C * HW));
    const int g = c / (C / GROUPS);
    const float m = stats[2 * (b * GROUPS + g)];
    const float r = stats[2 * (b * GROUPS + g) + 1];
    float v = (x[i] - m) * r * gam[c] + bet[c];
    if (do_silu) v = v / (1.f + __expf(-v));
    y[i] = v;
}

// ---------------------------------------------------------------------------
// Nearest-neighbor 2x upsample.
// ---------------------------------------------------------------------------
__global__ __launch_bounds__(256)
void upsample2x_kernel(const float* __restrict__ x, float* __restrict__ y,
                       int B, int C, int H, int W)
{
    const int OH = 2 * H, OW = 2 * W;
    const size_t total = (size_t)B * C * OH * OW;
    const size_t i = (size_t)blockIdx.x * 256 + threadIdx.x;
    if (i >= total) return;
    const int ow = (int)(i % OW);
    size_t t = i / OW;
    const int oh = (int)(t % OH); t /= OH;
    const int c  = (int)(t % C);
    const int b  = (int)(t / C);
    y[i] = x[(((size_t)b * C + c) * H + (oh >> 1)) * W + (ow >> 1)];
}

// ---------------------------------------------------------------------------
// Attention over flattened 8x8 spatial (HW<=64), one block per (query i, batch).
// ---------------------------------------------------------------------------
__global__ __launch_bounds__(256)
void attn_rows_kernel(const float* __restrict__ q, const float* __restrict__ k,
                      const float* __restrict__ v, float* __restrict__ out,
                      int C, int HW, float scale)
{
    const int i = blockIdx.x;
    const int b = blockIdx.y;
    __shared__ float sc[64];
    const size_t bo = (size_t)b * C * HW;

    for (int j = threadIdx.x; j < HW; j += 256) {
        float d = 0.f;
        for (int c = 0; c < C; ++c)
            d += q[bo + (size_t)c * HW + i] * k[bo + (size_t)c * HW + j];
        sc[j] = d * scale;
    }
    __syncthreads();
    if (threadIdx.x == 0) {
        float mx = -1e30f;
        for (int j = 0; j < HW; ++j) mx = fmaxf(mx, sc[j]);
        float s = 0.f;
        for (int j = 0; j < HW; ++j) { float e = __expf(sc[j] - mx); sc[j] = e; s += e; }
        const float inv = 1.f / s;
        for (int j = 0; j < HW; ++j) sc[j] *= inv;
    }
    __syncthreads();
    for (int c = threadIdx.x; c < C; c += 256) {
        float a = 0.f;
        for (int j = 0; j < HW; ++j)
            a += sc[j] * v[bo + (size_t)c * HW + j];
        out[bo + (size_t)c * HW + i] = a;
    }
}

// ---------------------------------------------------------------------------
// VQ: per-row argmin over 16384 codes (tie -> lowest index, matching argmin).
// ---------------------------------------------------------------------------
__global__ __launch_bounds__(256)
void vq_argmin_kernel(const float* __restrict__ z, const float* __restrict__ cb,
                      int* __restrict__ idx, int C, int HW, int K)
{
    const int row = blockIdx.x;     // b*HW + s
    const int b = row / HW, s = row % HW;
    __shared__ float zrow[256];
    __shared__ float bd[256];
    __shared__ int   bi[256];

    for (int c = threadIdx.x; c < C; c += 256)
        zrow[c] = z[((size_t)b * C + c) * HW + s];
    __syncthreads();

    float best = 3.4e38f; int besti = 0x7fffffff;
    for (int kk = threadIdx.x; kk < K; kk += 256) {
        const float* cp = cb + (size_t)kk * C;
        float d = 0.f;
        for (int c = 0; c < C; ++c) { float t = zrow[c] - cp[c]; d += t * t; }
        if (d < best || (d == best && kk < besti)) { best = d; besti = kk; }
    }
    bd[threadIdx.x] = best; bi[threadIdx.x] = besti;
    __syncthreads();
    for (int st = 128; st > 0; st >>= 1) {
        if (threadIdx.x < st) {
            const float od = bd[threadIdx.x + st];
            const int   oi = bi[threadIdx.x + st];
            if (od < bd[threadIdx.x] || (od == bd[threadIdx.x] && oi < bi[threadIdx.x])) {
                bd[threadIdx.x] = od; bi[threadIdx.x] = oi;
            }
        }
        __syncthreads();
    }
    if (threadIdx.x == 0) idx[row] = bi[0];
}

__global__ __launch_bounds__(256)
void vq_gather_loss_kernel(const float* __restrict__ z, const float* __restrict__ cb,
                           const int* __restrict__ idx, float* __restrict__ zq,
                           float* __restrict__ loss, int B, int C, int HW, float scale)
{
    const size_t total = (size_t)B * C * HW;
    const size_t i = (size_t)blockIdx.x * 256 + threadIdx.x;
    float d2 = 0.f;
    if (i < total) {
        const int s = (int)(i % HW);
        const int c = (int)((i / HW) % C);
        const int b = (int)(i / ((size_t)C * HW));
        const float qv = cb[(size_t)idx[b * HW + s] * C + c];
        const float d  = qv - z[i];
        zq[i] = qv;                 // straight-through: numeric value == zq
        d2 = d * d;
    }
    __shared__ float sh[256];
    sh[threadIdx.x] = d2; __syncthreads();
    for (int st = 128; st > 0; st >>= 1) {
        if (threadIdx.x < st) sh[threadIdx.x] += sh[threadIdx.x + st];
        __syncthreads();
    }
    if (threadIdx.x == 0) atomicAdd(loss, sh[0] * scale);
}

__global__ void zero_kernel(float* p) { *p = 0.f; }

// ===========================================================================
// Host orchestration
// ===========================================================================
static inline int cdiv(int a, int b) { return (a + b - 1) / b; }

struct ResP  { const float *g1,*b1,*w1,*c1,*g2,*b2,*w2,*c2,*ws,*bs; int cin,cout; };
struct AttnP { const float *g,*b,*wq,*bq,*wk,*bk,*wv,*bv,*wo,*bo; };
struct Blk   { int type; ResP r; const float *w, *bb; };   // 0=res, 1=down, 2=up

static void run_conv(hipStream_t s, const float* in, const float* w, const float* bias,
                     float* out, const float* resid, __bf16* wpk,
                     int B, int Cin, int H, int W, int Cout, int KH, int stride, int pad)
{
    const int KK = KH * KH;
    const size_t wtot = (size_t)Cout * Cin * KK;
    repack_w_kernel<<<(int)((wtot + 255) / 256), 256, 0, s>>>(w, wpk, Cout, Cin, KK);

    const int OH = (H + 2 * pad - KH) / stride + 1;
    const int OW = (W + 2 * pad - KH) / stride + 1;
    dim3 grid(cdiv(Cout, 32), cdiv(B * OH * OW, 32));
    if (KH == 3 && stride == 1)
        conv_wmma_kernel<3,3,1,1><<<grid,32,0,s>>>(in,wpk,bias,resid,out,B,Cin,H,W,Cout,OH,OW);
    else if (KH == 3)
        conv_wmma_kernel<3,3,2,1><<<grid,32,0,s>>>(in,wpk,bias,resid,out,B,Cin,H,W,Cout,OH,OW);
    else
        conv_wmma_kernel<1,1,1,0><<<grid,32,0,s>>>(in,wpk,bias,resid,out,B,Cin,H,W,Cout,OH,OW);
}

static void run_gn(hipStream_t s, const float* x, const float* g, const float* b,
                   float* y, float* stats, int B, int C, int HW, int silu)
{
    gn_stats_kernel<<<B * GROUPS, 256, 0, s>>>(x, stats, C, HW);
    const size_t tot = (size_t)B * C * HW;
    gn_apply_kernel<<<(int)((tot + 255) / 256), 256, 0, s>>>(x, stats, g, b, y, B, C, HW, silu);
}

static void run_res(hipStream_t s, const ResP& p, const float* in, float* out,
                    float* t1, float* t2, float* tsc, float* stats, __bf16* wpk,
                    int B, int H, int W)
{
    run_gn(s, in, p.g1, p.b1, t1, stats, B, p.cin, H * W, 1);
    run_conv(s, t1, p.w1, p.c1, t2, nullptr, wpk, B, p.cin, H, W, p.cout, 3, 1, 1);
    run_gn(s, t2, p.g2, p.b2, t1, stats, B, p.cout, H * W, 1);
    const float* sc = in;
    if (p.ws) { run_conv(s, in, p.ws, p.bs, tsc, nullptr, wpk, B, p.cin, H, W, p.cout, 1, 1, 0); sc = tsc; }
    run_conv(s, t1, p.w2, p.c2, out, sc, wpk, B, p.cout, H, W, p.cout, 3, 1, 1);
}

static void run_attn(hipStream_t s, const AttnP& a, const float* in, float* out,
                     float* t1, float* q, float* k, float* v, float* hf, float* stats,
                     __bf16* wpk, int B, int C, int H, int W)
{
    run_gn(s, in, a.g, a.b, t1, stats, B, C, H * W, 0);
    run_conv(s, t1, a.wq, a.bq, q, nullptr, wpk, B, C, H, W, C, 1, 1, 0);
    run_conv(s, t1, a.wk, a.bk, k, nullptr, wpk, B, C, H, W, C, 1, 1, 0);
    run_conv(s, t1, a.wv, a.bv, v, nullptr, wpk, B, C, H, W, C, 1, 1, 0);
    attn_rows_kernel<<<dim3(H * W, B), 256, 0, s>>>(q, k, v, hf, C, H * W, 1.0f / sqrtf((float)C));
    run_conv(s, hf, a.wo, a.bo, out, in, wpk, B, C, H, W, C, 1, 1, 0);
}

extern "C" void kernel_launch(void* const* d_in, const int* in_sizes, int n_in,
                              void* d_out, int out_size, void* d_ws, size_t ws_size,
                              hipStream_t stream)
{
    (void)in_sizes; (void)n_in; (void)ws_size;
    const int B = 2;

    // ---- parse d_in in setup_inputs() insertion order ----
    int cur_idx = 0;
    auto nxt = [&]() { return (const float*)d_in[cur_idx++]; };
    auto parse_res = [&](int cin, int cout) {
        ResP p{}; p.cin = cin; p.cout = cout;
        p.g1 = nxt(); p.b1 = nxt(); p.w1 = nxt(); p.c1 = nxt();
        p.g2 = nxt(); p.b2 = nxt(); p.w2 = nxt(); p.c2 = nxt();
        if (cin != cout) { p.ws = nxt(); p.bs = nxt(); } else { p.ws = nullptr; p.bs = nullptr; }
        return p;
    };
    auto parse_attn = [&]() {
        AttnP a{};
        a.g = nxt(); a.b = nxt(); a.wq = nxt(); a.bq = nxt(); a.wk = nxt(); a.bk = nxt();
        a.wv = nxt(); a.bv = nxt(); a.wo = nxt(); a.bo = nxt();
        return a;
    };

    const float* x_in = nxt();                         // 'x'
    // params.enc
    const float *e_ciw = nxt(), *e_cib = nxt();
    Blk eblk[14]; int ne = 0; int cur = 128;
    const int mult[5] = {1, 1, 2, 2, 4};
    for (int i = 0; i < 5; ++i) {
        const int co = 128 * mult[i];
        for (int rr = 0; rr < 2; ++rr) { eblk[ne].type = 0; eblk[ne].r = parse_res(cur, co); ++ne; cur = co; }
        if (i < 4) { eblk[ne].type = 1; eblk[ne].w = nxt(); eblk[ne].bb = nxt(); ++ne; }
    }
    ResP  e_mid1 = parse_res(cur, cur);
    AttnP e_attn = parse_attn();
    ResP  e_mid2 = parse_res(cur, cur);
    const float *e_gg = nxt(), *e_gb = nxt(), *e_cow = nxt(), *e_cob = nxt();
    // params.dec
    const float *d_ciw = nxt(), *d_cib = nxt();
    ResP  d_mid1 = parse_res(512, 512);
    AttnP d_attn = parse_attn();
    ResP  d_mid2 = parse_res(512, 512);
    Blk dblk[19]; int nd = 0; cur = 512;
    const int rmult[5] = {4, 2, 2, 1, 1};
    for (int i = 0; i < 5; ++i) {
        const int co = 128 * rmult[i];
        for (int rr = 0; rr < 3; ++rr) { dblk[nd].type = 0; dblk[nd].r = parse_res(cur, co); ++nd; cur = co; }
        if (i < 4) { dblk[nd].type = 2; dblk[nd].w = nxt(); dblk[nd].bb = nxt(); ++nd; }
    }
    const float *d_gg = nxt(), *d_gb = nxt(), *d_cow = nxt(), *d_cob = nxt();
    // params.codebook, quant_conv, post_quant_conv
    const float *codebook = nxt();
    const float *qcw = nxt(), *qcb = nxt(), *pcw = nxt(), *pcb = nxt();

    // ---- workspace bump allocator ----
    size_t off = 0;
    auto alloc = [&](size_t bytes) {
        void* p = (char*)d_ws + off;
        off += ((bytes + 255) & ~(size_t)255);
        return p;
    };
    const size_t BUFE = (size_t)2 * 128 * 128 * 128;   // largest activation (elements)
    float *bA = (float*)alloc(BUFE * 4), *bB = (float*)alloc(BUFE * 4);
    float *t1 = (float*)alloc(BUFE * 4), *t2 = (float*)alloc(BUFE * 4), *t3 = (float*)alloc(BUFE * 4);
    float *stats = (float*)alloc(2 * B * GROUPS * 4);
    float *aq = (float*)alloc(2 * 512 * 64 * 4), *ak = (float*)alloc(2 * 512 * 64 * 4);
    float *av = (float*)alloc(2 * 512 * 64 * 4), *ah = (float*)alloc(2 * 512 * 64 * 4);
    int   *idxb = (int*)alloc(2 * 64 * 4);
    __bf16 *wpk = (__bf16*)alloc((size_t)512 * 512 * 9 * 2);   // packed-weight scratch (max conv)

    float* loss_p = (float*)d_out + (out_size - 1);
    zero_kernel<<<1, 1, 0, stream>>>(loss_p);

    auto swapAB = [&]() { float* t = bA; bA = bB; bB = t; };

    // ================= encoder =================
    int Hc = 128, cch = 128;
    run_conv(stream, x_in, e_ciw, e_cib, bA, nullptr, wpk, B, 3, Hc, Hc, 128, 3, 1, 1);
    for (int ib = 0; ib < ne; ++ib) {
        if (eblk[ib].type == 0) {
            run_res(stream, eblk[ib].r, bA, bB, t1, t2, t3, stats, wpk, B, Hc, Hc);
            swapAB(); cch = eblk[ib].r.cout;
        } else {
            run_conv(stream, bA, eblk[ib].w, eblk[ib].bb, bB, nullptr, wpk, B, cch, Hc, Hc, cch, 3, 2, 1);
            swapAB(); Hc /= 2;
        }
    }
    run_res(stream, e_mid1, bA, bB, t1, t2, t3, stats, wpk, B, Hc, Hc); swapAB();
    run_attn(stream, e_attn, bA, bB, t1, aq, ak, av, ah, stats, wpk, B, 512, Hc, Hc); swapAB();
    run_res(stream, e_mid2, bA, bB, t1, t2, t3, stats, wpk, B, Hc, Hc); swapAB();
    run_gn(stream, bA, e_gg, e_gb, t1, stats, B, 512, Hc * Hc, 1);
    run_conv(stream, t1, e_cow, e_cob, bB, nullptr, wpk, B, 512, Hc, Hc, 256, 3, 1, 1); swapAB();

    // quant conv (1x1)
    run_conv(stream, bA, qcw, qcb, bB, nullptr, wpk, B, 256, Hc, Hc, 256, 1, 1, 0); swapAB();

    // ================= quantize =================
    const int HWz = Hc * Hc;                        // 64
    vq_argmin_kernel<<<B * HWz, 256, 0, stream>>>(bA, codebook, idxb, 256, HWz, 16384);
    const size_t ztot = (size_t)B * 256 * HWz;
    vq_gather_loss_kernel<<<(int)((ztot + 255) / 256), 256, 0, stream>>>(
        bA, codebook, idxb, bB, loss_p, B, 256, HWz, 1.25f / (float)ztot);
    swapAB();

    // post-quant conv (1x1)
    run_conv(stream, bA, pcw, pcb, bB, nullptr, wpk, B, 256, Hc, Hc, 256, 1, 1, 0); swapAB();

    // ================= decoder =================
    run_conv(stream, bA, d_ciw, d_cib, bB, nullptr, wpk, B, 256, Hc, Hc, 512, 3, 1, 1); swapAB();
    cch = 512;
    run_res(stream, d_mid1, bA, bB, t1, t2, t3, stats, wpk, B, Hc, Hc); swapAB();
    run_attn(stream, d_attn, bA, bB, t1, aq, ak, av, ah, stats, wpk, B, 512, Hc, Hc); swapAB();
    run_res(stream, d_mid2, bA, bB, t1, t2, t3, stats, wpk, B, Hc, Hc); swapAB();
    for (int ib = 0; ib < nd; ++ib) {
        if (dblk[ib].type == 0) {
            run_res(stream, dblk[ib].r, bA, bB, t1, t2, t3, stats, wpk, B, Hc, Hc);
            swapAB(); cch = dblk[ib].r.cout;
        } else {
            const size_t utot = (size_t)B * cch * (2 * Hc) * (2 * Hc);
            upsample2x_kernel<<<(int)((utot + 255) / 256), 256, 0, stream>>>(bA, t1, B, cch, Hc, Hc);
            Hc *= 2;
            run_conv(stream, t1, dblk[ib].w, dblk[ib].bb, bB, nullptr, wpk, B, cch, Hc, Hc, cch, 3, 1, 1);
            swapAB();
        }
    }
    run_gn(stream, bA, d_gg, d_gb, t1, stats, B, cch, Hc * Hc, 1);
    run_conv(stream, t1, d_cow, d_cob, (float*)d_out, nullptr, wpk, B, cch, Hc, Hc, 3, 3, 1, 1);
}
